// SageConv_76476187673102
// MI455X (gfx1250) — matrix operand order repaired and checked by reading, not compiled, not run on hardware
//
#include <hip/hip_runtime.h>

#define N_NODES 10000
#define F_IN    256
#define F_OUT   256
#define K_TOT   512   // 2 * F_IN

typedef __attribute__((ext_vector_type(16))) _Float16 v16h;
typedef __attribute__((ext_vector_type(8)))  float    v8f;

// ---------------------------------------------------------------- zero scratch
__global__ void sage_zero_kernel(float* __restrict__ ws, int n) {
    int i = blockIdx.x * blockDim.x + threadIdx.x;
    if (i < n) ws[i] = 0.0f;
}

// ---------------------------------------------------------------- edge scatter
// 64 threads per edge; each thread handles 4 contiguous floats (float4 load +
// 4x global_atomic_add_f32). A wave32 covers 512 contiguous bytes of one row.
__global__ void sage_scatter_kernel(const float* __restrict__ h,
                                    const int*   __restrict__ ei,
                                    float* __restrict__ msg,
                                    float* __restrict__ deg,
                                    int n_edges) {
    long long t = (long long)blockIdx.x * blockDim.x + threadIdx.x;
    int e = (int)(t >> 6);
    if (e >= n_edges) return;
    int c = ((int)t & 63) * 4;

    int src = ei[e];
    int dst = ei[n_edges + e];

    const float4 v = *reinterpret_cast<const float4*>(h + (long long)src * F_IN + c);
    float* p = msg + (long long)dst * F_IN + c;
    atomicAdd(p + 0, v.x);
    atomicAdd(p + 1, v.y);
    atomicAdd(p + 2, v.z);
    atomicAdd(p + 3, v.w);
    if (c == 0) atomicAdd(deg + dst, 1.0f);
}

// ---------------------------------------------------------------- WMMA GEMM
// A-fragment builder: 16x32 f16 layout (ISA 7.12.2). akb = 0 (lanes 0-15) or
// 8 (lanes 16-31); halves 0-7 = K akb..akb+7, halves 8-15 = K akb+16..akb+23.
__device__ inline v16h sage_load_a(const float* __restrict__ arow, int akb, float s) {
    float4 a0 = *reinterpret_cast<const float4*>(arow + akb);
    float4 a1 = *reinterpret_cast<const float4*>(arow + akb + 4);
    float4 a2 = *reinterpret_cast<const float4*>(arow + akb + 16);
    float4 a3 = *reinterpret_cast<const float4*>(arow + akb + 20);
    v16h a;
    a[0]  = (_Float16)(a0.x * s); a[1]  = (_Float16)(a0.y * s);
    a[2]  = (_Float16)(a0.z * s); a[3]  = (_Float16)(a0.w * s);
    a[4]  = (_Float16)(a1.x * s); a[5]  = (_Float16)(a1.y * s);
    a[6]  = (_Float16)(a1.z * s); a[7]  = (_Float16)(a1.w * s);
    a[8]  = (_Float16)(a2.x * s); a[9]  = (_Float16)(a2.y * s);
    a[10] = (_Float16)(a2.z * s); a[11] = (_Float16)(a2.w * s);
    a[12] = (_Float16)(a3.x * s); a[13] = (_Float16)(a3.y * s);
    a[14] = (_Float16)(a3.z * s); a[15] = (_Float16)(a3.w * s);
    return a;
}

// B-fragment: 32x16 f16. Lane = column n; halves 0-7 = K bkb..bkb+7, halves 8-15 =
// K bkb+16..bkb+23, bkb = 0 (lanes 0-15) or 8 (lanes 16-31). B[k][n] = W[n][k] (W row-major).
__device__ inline v16h sage_load_b(const float* __restrict__ wrow, int bkb) {
    float4 b0 = *reinterpret_cast<const float4*>(wrow + bkb);
    float4 b1 = *reinterpret_cast<const float4*>(wrow + bkb + 4);
    float4 b2 = *reinterpret_cast<const float4*>(wrow + bkb + 16);
    float4 b3 = *reinterpret_cast<const float4*>(wrow + bkb + 20);
    v16h bb;
    bb[0]  = (_Float16)b0.x; bb[1]  = (_Float16)b0.y;
    bb[2]  = (_Float16)b0.z; bb[3]  = (_Float16)b0.w;
    bb[4]  = (_Float16)b1.x; bb[5]  = (_Float16)b1.y;
    bb[6]  = (_Float16)b1.z; bb[7]  = (_Float16)b1.w;
    bb[8]  = (_Float16)b2.x; bb[9]  = (_Float16)b2.y;
    bb[10] = (_Float16)b2.z; bb[11] = (_Float16)b2.w;
    bb[12] = (_Float16)b3.x; bb[13] = (_Float16)b3.y;
    bb[14] = (_Float16)b3.z; bb[15] = (_Float16)b3.w;
    return bb;
}

__global__ __launch_bounds__(256) void sage_gemm_kernel(
        const float* __restrict__ h,
        const float* __restrict__ msg,
        const float* __restrict__ deg,
        const float* __restrict__ W,
        const float* __restrict__ b,
        float* __restrict__ out) {
    const int wave  = blockIdx.x * (blockDim.x >> 5) + (threadIdx.x >> 5);
    const int lane  = threadIdx.x & 31;
    const int mtile = wave >> 4;   // 0..624  (10000/16 = 625 exactly)
    const int ntile = wave & 15;   // 0..15   (256/16)
    if (mtile >= N_NODES / 16) return;

    const int half = lane >> 4;    // 0 or 1
    const int l    = lane & 15;

    const int node = mtile * 16 + l;          // A row (node index)
    const int ocol = ntile * 16 + l;          // B column (output feature)

    const float dg   = deg[node];
    const float invd = 1.0f / fmaxf(dg, 1.0f);

    const int akb = half ? 8 : 0;
    const int bkb = half ? 8 : 0;

    const float* hrow = h   + (long long)node * F_IN;
    const float* mrow = msg + (long long)node * F_IN;
    const float* wrow = W   + (long long)ocol * K_TOT;

    v8f c = {};
    // K = 0..255: A from h (scale 1)
    for (int kt = 0; kt < F_IN / 32; ++kt) {
        v16h a  = sage_load_a(hrow + kt * 32, akb, 1.0f);
        v16h bb = sage_load_b(wrow + kt * 32, bkb);
        c = __builtin_amdgcn_wmma_f32_16x16x32_f16(false, a, false, bb,
                                                   (short)0, c, false, false);
    }
    // K = 256..511: A from msg_sum * (1/max(deg,1))
    for (int kt = 0; kt < F_IN / 32; ++kt) {
        v16h a  = sage_load_a(mrow + kt * 32, akb, invd);
        v16h bb = sage_load_b(wrow + F_IN + kt * 32, bkb);
        c = __builtin_amdgcn_wmma_f32_16x16x32_f16(false, a, false, bb,
                                                   (short)0, c, false, false);
    }

    // D layout: VGPR r -> M = r + 8*half (lanes 16-31 hold M=8..15), N = l.
    const float bias = b[ocol];
    #pragma unroll
    for (int r = 0; r < 8; ++r) {
        int m = mtile * 16 + r + 8 * half;
        out[(long long)m * F_OUT + ocol] = c[r] + bias;
    }
}

// ---------------------------------------------------------------- launch
extern "C" void kernel_launch(void* const* d_in, const int* in_sizes, int n_in,
                              void* d_out, int out_size, void* d_ws, size_t ws_size,
                              hipStream_t stream) {
    const float* h  = (const float*)d_in[0];
    const int*   ei = (const int*)d_in[1];
    const float* W  = (const float*)d_in[2];
    const float* b  = (const float*)d_in[3];
    float* out = (float*)d_out;

    const int n_edges = in_sizes[1] / 2;

    float* msg = (float*)d_ws;                       // [N_NODES, F_IN]
    float* deg = msg + (size_t)N_NODES * F_IN;       // [N_NODES]

    // 1) zero msg_sum + deg (must happen every call; ws is not re-poisoned)
    const int nz = N_NODES * F_IN + N_NODES;
    sage_zero_kernel<<<(nz + 255) / 256, 256, 0, stream>>>((float*)d_ws, nz);

    // 2) gather h[src] -> atomic accumulate into msg[dst], count degree
    const long long sthreads = (long long)n_edges * 64;
    sage_scatter_kernel<<<(unsigned)((sthreads + 255) / 256), 256, 0, stream>>>(
        h, ei, msg, deg, n_edges);

    // 3) fused mean + concat + linear via WMMA: one wave per 16x16 out tile
    const int n_waves  = (N_NODES / 16) * (F_OUT / 16);   // 10000
    const int n_blocks = n_waves / 8;                     // 8 waves / 256-thr block
    sage_gemm_kernel<<<n_blocks, 256, 0, stream>>>(h, msg, deg, W, b, out);
}
